// ContractiveREN_1047972020310
// MI455X (gfx1250) — matrix operand order repaired and checked by reading, not compiled, not run on hardware
//
#include <hip/hip_runtime.h>

// ---------------------------------------------------------------------------
// Contractive REN on MI455X (gfx1250, wave32).
// 16 workgroups x 16 batch rows; whole T-loop in-kernel; weights in LDS (bf16
// fragment-swizzled for v_wmma_f32_16x16x32_bf16); f32 forward substitution.
// CDNA5 async engine: prologue weight streaming + double-buffered u_t staging
// via global_load_async_to_lds_b128 / s_wait_asynccnt.
// ---------------------------------------------------------------------------

#define REN_EPS 0.001f

typedef __attribute__((ext_vector_type(16))) __bf16       v16bf;
typedef __attribute__((ext_vector_type(8)))  float        v8f;
typedef __attribute__((ext_vector_type(4)))  unsigned int v4u;
typedef __attribute__((ext_vector_type(2)))  unsigned int v2u;

__device__ __forceinline__ unsigned short f2bf(float f) {
  unsigned u = __float_as_uint(f);
  u += 0x7FFFu + ((u >> 16) & 1u);          // round-to-nearest-even
  return (unsigned short)(u >> 16);
}

// Async copy of 16B from global to LDS (ASYNCcnt-tracked, no VGPR data path).
// lds_off is the byte offset inside the (purely dynamic) LDS allocation.
__device__ __forceinline__ void async_b128(unsigned lds_off, const void* g) {
  unsigned long long ga = (unsigned long long)(uintptr_t)g;
  asm volatile("global_load_async_to_lds_b128 %0, %1, off"
               :: "v"(lds_off), "v"(ga) : "memory");
}

// 16 bf16 elements from two 16B LDS chunks -> WMMA operand register file.
__device__ __forceinline__ v16bf load_frag(const unsigned short* p0,
                                           const unsigned short* p1) {
  union { v4u u[2]; v16bf v; } f;
  f.u[0] = *(const v4u*)p0;
  f.u[1] = *(const v4u*)p1;
  return f.v;
}

// One 16x16 output tile, K = 32*KB.  A: row-major bf16 [16][RS] in LDS.
// B: fragment-swizzled weight (lane-contiguous 16-element chunks).
__device__ __forceinline__ v8f gemm_tile(const unsigned short* A, int RS,
                                         const unsigned short* W, int KB,
                                         int nb, int lanelo, int laneg,
                                         int lane, v8f c) {
  for (int kb = 0; kb < KB; ++kb) {
    const unsigned short* a0 = A + lanelo * RS + kb * 32 + 16 * laneg;
    v16bf a = load_frag(a0, a0 + 8);
    const unsigned short* b0 = W + (((nb * KB + kb) * 32 + lane) << 4);
    v16bf b = load_frag(b0, b0 + 8);
    c = __builtin_amdgcn_wmma_f32_16x16x32_bf16(false, a, false, b,
                                                (short)0, c, false, false);
  }
  return c;
}

// ---------------- Phase 1a: H = X^T X + eps*I  (384x384) -------------------
__global__ void k_buildH(const float* __restrict__ X, float* __restrict__ H) {
  int idx = blockIdx.x * blockDim.x + threadIdx.x;
  if (idx >= 384 * 384) return;
  int r = idx / 384, c = idx % 384;
  float s = 0.f;
  for (int m = 0; m < 384; ++m) s = fmaf(X[m * 384 + r], X[m * 384 + c], s);
  if (r == c) s += REN_EPS;
  H[idx] = s;
}

// ---------------- Phase 1b: E = 0.5(H11 + H33 + Y - Y^T); Einv = E^-1 ------
// Gauss-Jordan without pivoting (E has PD symmetric part -> nonsingular
// leading minors).  One block, LDS-resident [E|I].
__global__ __launch_bounds__(256) void k_invE(const float* __restrict__ H,
                                              const float* __restrict__ Y,
                                              float* __restrict__ Einv) {
  extern __shared__ char sm2[];
  float* M    = (float*)sm2;              // 128 x 256
  float* colk = (float*)(sm2 + 131072);   // 128
  const int tid = threadIdx.x;
  for (int i = tid; i < 128 * 128; i += 256) {
    int r = i >> 7, c = i & 127;
    M[r * 256 + c] = 0.5f * (H[r * 384 + c] + H[(256 + r) * 384 + (256 + c)] +
                             Y[r * 128 + c] - Y[c * 128 + r]);
    M[r * 256 + 128 + c] = (r == c) ? 1.f : 0.f;
  }
  __syncthreads();
  for (int k = 0; k < 128; ++k) {
    float pinv = 1.f / M[k * 256 + k];
    M[k * 256 + tid] *= pinv;             // scale pivot row (one col/thread)
    __syncthreads();
    if (tid < 128) colk[tid] = (tid == k) ? 0.f : M[tid * 256 + k];
    __syncthreads();
    float pk = M[k * 256 + tid];          // thread owns column 'tid'
    for (int r = 0; r < 128; ++r)
      M[r * 256 + tid] = fmaf(-colk[r], pk, M[r * 256 + tid]);
    __syncthreads();
  }
  for (int i = tid; i < 128 * 128; i += 256) {
    int r = i >> 7, c = i & 127;
    Einv[i] = M[r * 256 + 128 + c];
  }
}

// ---------------- Phase 1c: pack derived weights ---------------------------
// Fragment swizzle: off = (((nb*KB + kb)*32 + lane)*16 + e)
//   k = kb*32 + (lane>>4)*16 + e ,  n = nb*16 + (lane&15) ,  value = W[k][n]
__global__ void k_pack(const float* __restrict__ H, const float* __restrict__ Einv,
                       const float* __restrict__ B2, const float* __restrict__ C2,
                       const float* __restrict__ D12, const float* __restrict__ x_init,
                       float* __restrict__ LamInv, float* __restrict__ y0v,
                       float* __restrict__ D11c, unsigned short* __restrict__ W1,
                       unsigned short* __restrict__ W2, unsigned short* __restrict__ WEm,
                       unsigned short* __restrict__ WCm) {
  int idx = blockIdx.x * blockDim.x + threadIdx.x;
  if (idx < 128) {
    LamInv[idx] = 2.f / H[(128 + idx) * 384 + (128 + idx)];   // 1/Lam
  } else if (idx < 192) {
    int n = idx - 128;                                        // y0 = C2 @ x0
    float s = 0.f;
    for (int k = 0; k < 128; ++k) s = fmaf(C2[n * 128 + k], x_init[k], s);
    y0v[n] = s;
  } else if (idx < 16576) {
    int q = idx - 192, j = q >> 7, i = q & 127;               // D11c[j][i]=D11[i][j]
    D11c[j * 128 + i] = (i > j) ? -H[(128 + i) * 384 + (128 + j)] : 0.f;
  } else if (idx < 41152) {                                   // W1=[C1^T;D12^T] K=192
    int off = idx - 16576;
    int e = off & 15, lane = (off >> 4) & 31, blk = off >> 9;
    int kb = blk % 6, nb = blk / 6;
    int k = kb * 32 + (lane >> 4) * 16 + e, n = nb * 16 + (lane & 15);
    float v = (k < 128) ? -H[(128 + n) * 384 + k] : D12[n * 64 + (k - 128)];
    W1[off] = f2bf(v);
  } else if (idx < 82112) {                                   // W2=[Fm^T;B2^T;B1^T] K=320
    int off = idx - 41152;
    int e = off & 15, lane = (off >> 4) & 31, blk = off >> 9;
    int kb = blk % 10, nb = blk / 10;
    int k = kb * 32 + (lane >> 4) * 16 + e, n = nb * 16 + (lane & 15);
    float v;
    if (k < 128)      v = H[(256 + n) * 384 + k];
    else if (k < 192) v = B2[n * 64 + (k - 128)];
    else              v = H[(256 + n) * 384 + 128 + (k - 192)];
    W2[off] = f2bf(v);
  } else if (idx < 98496) {                                   // WE = Einv^T K=128
    int off = idx - 82112;
    int e = off & 15, lane = (off >> 4) & 31, blk = off >> 9;
    int kb = blk % 4, nb = blk / 4;
    int k = kb * 32 + (lane >> 4) * 16 + e, n = nb * 16 + (lane & 15);
    WEm[off] = f2bf(Einv[n * 128 + k]);
  } else if (idx < 106688) {                                  // WC = C2^T K=128,N=64
    int off = idx - 98496;
    int e = off & 15, lane = (off >> 4) & 31, blk = off >> 9;
    int kb = blk % 4, nb = blk / 4;
    int k = kb * 32 + (lane >> 4) * 16 + e, n = nb * 16 + (lane & 15);
    WCm[off] = f2bf(C2[n * 128 + k]);
  }
}

// ---------------- Phase 2: the sequential scan -----------------------------
// LDS map (bytes) -- mirrors the flat ws region [D11c|LamInv|y0|W1|W2|WE|WC]:
//   d11c @0 (64K) | lamI @65536 (512) | y0 @66048 (256) | W1 @66304 (48K) |
//   W2 @115456 (80K) | WE @197376 (32K) | WC @230144 (16K) |
//   act @246528 (16x320 bf16) | tmpb @256768 (16x128 bf16) |
//   accl @260864 (16x128 f32) | wrow @269056 (2x16 f32) |
//   stage @269184 (2 x 4KB raw f32 u_t double buffer)  => 277376 total
__global__ __launch_bounds__(256) void ren_seq_kernel(
    const float* __restrict__ u_in, const float* __restrict__ x_init,
    const char* __restrict__ wsFlat,   // 246528 B: D11c|LamInv|y0|W1|W2|WE|WC
    float* __restrict__ out) {
  extern __shared__ char smem[];
  float*          d11c = (float*)(smem);
  float*          lamI = (float*)(smem + 65536);
  float*          y0l  = (float*)(smem + 66048);
  unsigned short* W1   = (unsigned short*)(smem + 66304);
  unsigned short* W2   = (unsigned short*)(smem + 115456);
  unsigned short* WEm  = (unsigned short*)(smem + 197376);
  unsigned short* WCm  = (unsigned short*)(smem + 230144);
  unsigned short* act  = (unsigned short*)(smem + 246528);
  unsigned short* tmpb = (unsigned short*)(smem + 256768);
  float*          accl = (float*)(smem + 260864);
  float*          wrow = (float*)(smem + 269056);
  const unsigned  stage0 = 269184;           // byte offsets of u stage buffers

  const int tid = threadIdx.x;
  const int wv = tid >> 5, lane = tid & 31;
  const int lanelo = lane & 15, laneg = lane >> 4;
  const int ncol = lanelo + 16 * wv;
  const int b0 = blockIdx.x * 16;
  const int sb = tid & 15, ib = tid >> 4, i0 = ib * 8;   // w-solve ownership
  const int um = tid >> 4, ukq = (tid & 15) << 2;        // u staging mapping

  // ---- prologue ----
  // 1) stream all derived constants into LDS via the async engine (246528 B
  //    = 15408 x b128, no VGPR data path, ASYNCcnt-tracked)
  for (int i = tid; i < 15408; i += 256)
    async_b128((unsigned)(i * 16), wsFlat + i * 16);
  // 2) stage u_0 asynchronously (each thread owns 16 bytes, row-major [16][64])
  async_b128(stage0 + tid * 16,
             u_in + ((b0 + um) * 512 + 0) * 64 + ukq);
  // 3) init activations: x0 (bf16) | u zeros | w zeros
  for (int i = tid; i < 16 * 320; i += 256) {
    int k = i % 320;
    act[i] = (k < 128) ? f2bf(x_init[k]) : (unsigned short)0;
  }
  asm volatile("s_wait_asynccnt 0x0" ::: "memory");
  __syncthreads();
  // 4) emit y[:,0,:] = C2 @ x0 from the LDS-resident y0 vector
  for (int i = tid; i < 1024; i += 256) {
    int m = i >> 6, n = i & 63;
    out[(b0 + m) * (512 * 64) + n] = y0l[n];
  }

  for (int t = 0; t < 511; ++t) {
    // ---- 1. issue async stage of u_{t+1}; convert u_t from LDS stage ----
    if (t + 1 < 511) {
      async_b128(stage0 + ((t + 1) & 1) * 4096 + tid * 16,
                 u_in + ((b0 + um) * 512 + (t + 1)) * 64 + ukq);
      // in-order async loads: <=1 outstanding  =>  u_t stage is resident
      asm volatile("s_wait_asynccnt 0x1" ::: "memory");
    } else {
      asm volatile("s_wait_asynccnt 0x0" ::: "memory");
    }
    {
      const float* sp = (const float*)(smem + stage0 + (t & 1) * 4096) +
                        um * 64 + ukq;
      float4 uv = *(const float4*)sp;
      v2u pk;
      pk.x = (unsigned)f2bf(uv.x) | ((unsigned)f2bf(uv.y) << 16);
      pk.y = (unsigned)f2bf(uv.z) | ((unsigned)f2bf(uv.w) << 16);
      *(v2u*)(act + um * 320 + 128 + ukq) = pk;
    }
    __syncthreads();

    // ---- 2. vx = [x|u] @ W1  (K=192) ----
    {
      v8f c = {0.f, 0.f, 0.f, 0.f, 0.f, 0.f, 0.f, 0.f};
      c = gemm_tile(act, 320, W1, 6, wv, lanelo, laneg, lane, c);
      for (int r = 0; r < 8; ++r) accl[(r + 8 * laneg) * 128 + ncol] = c[r];
    }
    __syncthreads();

    // ---- 3. forward substitution: w[:,j] = tanh((vx + w@D11[j]) / Lam[j]) --
    // Thread (sb, ib) owns batch row sb, i in [8*ib, 8*ib+8); acc in regs.
    // D11c[j][i] == 0 for i <= j -> unconditional rank-1 update.
    {
      float a8[8];
#pragma unroll
      for (int r = 0; r < 8; ++r) a8[r] = accl[sb * 128 + i0 + r];
      for (int j = 0; j < 128; ++j) {
        if ((j >> 3) == ib) {
          float wj = tanhf(a8[j & 7] * lamI[j]);
          wrow[(j & 1) * 16 + sb] = wj;
          act[sb * 320 + 192 + j] = f2bf(wj);
        }
        __syncthreads();
        float wb = wrow[(j & 1) * 16 + sb];
        const float* dr = d11c + j * 128 + i0;
#pragma unroll
        for (int r = 0; r < 8; ++r) a8[r] = fmaf(wb, dr[r], a8[r]);
      }
    }

    // ---- 4. acc2 = [x|u|w] @ W2  (K=320) -> tmpb (bf16) ----
    {
      v8f c = {0.f, 0.f, 0.f, 0.f, 0.f, 0.f, 0.f, 0.f};
      c = gemm_tile(act, 320, W2, 10, wv, lanelo, laneg, lane, c);
      for (int r = 0; r < 8; ++r)
        tmpb[(r + 8 * laneg) * 128 + ncol] = f2bf(c[r]);
    }
    __syncthreads();

    // ---- 5. x_new = acc2 @ Einv^T -> act[:,0:128] (bf16 state) ----
    {
      v8f c = {0.f, 0.f, 0.f, 0.f, 0.f, 0.f, 0.f, 0.f};
      c = gemm_tile(tmpb, 128, WEm, 4, wv, lanelo, laneg, lane, c);
      for (int r = 0; r < 8; ++r)
        act[(r + 8 * laneg) * 320 + ncol] = f2bf(c[r]);
    }
    __syncthreads();

    // ---- 6. y_{t+1} = x_new @ C2^T  (waves 0..3, wave-uniform branch) ----
    if (wv < 4) {
      v8f c = {0.f, 0.f, 0.f, 0.f, 0.f, 0.f, 0.f, 0.f};
      c = gemm_tile(act, 320, WCm, 4, wv, lanelo, laneg, lane, c);
      for (int r = 0; r < 8; ++r)
        out[((b0 + r + 8 * laneg) * 512 + (t + 1)) * 64 + ncol] = c[r];
    }
    __syncthreads();
  }
}

// ---------------------------------------------------------------------------
extern "C" void kernel_launch(void* const* d_in, const int* in_sizes, int n_in,
                              void* d_out, int out_size, void* d_ws, size_t ws_size,
                              hipStream_t stream) {
  const float* u_in   = (const float*)d_in[0];
  const float* X      = (const float*)d_in[1];
  const float* Y      = (const float*)d_in[2];
  const float* B2     = (const float*)d_in[3];
  const float* C2     = (const float*)d_in[4];
  const float* D12    = (const float*)d_in[5];
  const float* x_init = (const float*)d_in[6];
  float* out = (float*)d_out;
  char* ws = (char*)d_ws;

  // ws layout (flat region 655360..901888 is async-copied to LDS in one pass)
  float* H      = (float*)(ws);               // 384*384 f32
  float* Einv   = (float*)(ws + 589824);      // 128*128 f32
  float* D11c   = (float*)(ws + 655360);      // 128*128 f32
  float* LamInv = (float*)(ws + 720896);      // 128 f32
  float* y0v    = (float*)(ws + 721408);      // 64 f32
  unsigned short* W1  = (unsigned short*)(ws + 721664);   // 192*128 bf16
  unsigned short* W2  = (unsigned short*)(ws + 770816);   // 320*128 bf16
  unsigned short* WEm = (unsigned short*)(ws + 852736);   // 128*128 bf16
  unsigned short* WCm = (unsigned short*)(ws + 885504);   // 128*64  bf16

  hipFuncSetAttribute((const void*)k_invE,
                      hipFuncAttributeMaxDynamicSharedMemorySize, 132096);
  hipFuncSetAttribute((const void*)ren_seq_kernel,
                      hipFuncAttributeMaxDynamicSharedMemorySize, 277376);

  k_buildH<<<576, 256, 0, stream>>>(X, H);
  k_invE<<<1, 256, 132096, stream>>>(H, Y, Einv);
  k_pack<<<417, 256, 0, stream>>>(H, Einv, B2, C2, D12, x_init,
                                  LamInv, y0v, D11c, W1, W2, WEm, WCm);
  ren_seq_kernel<<<16, 256, 277376, stream>>>(u_in, x_init,
                                              (const char*)(ws + 655360), out);
}
